// LocalDeepRFMFull_20066087207285
// MI455X (gfx1250) — compile-verified, matrix operand order checked by reading, not run on hardware
//
#include <hip/hip_runtime.h>
#include <hip/hip_bf16.h>

// ---------------------------------------------------------------------------
// LocalDeepRFM on MI455X (gfx1250). One 256-thread workgroup (8 x wave32)
// owns 16 sample rows x all 128 groups x all 4 stages. The whole recurrence
// lives in LDS (x tile + double-buffered z tile); all matmuls run through
// v_wmma_f32_16x16x32_bf16 with f32 accumulation. WMMA A-fragments for the
// input GEMM are loaded straight out of the resident x/z tiles (every
// 8-element K-run of the fragment layout stays inside one source window and
// never crosses the mod-1024 wrap).
// ---------------------------------------------------------------------------

typedef __attribute__((ext_vector_type(16))) __bf16 v16bf;
typedef __attribute__((ext_vector_type(8)))  __bf16 v8bf;
typedef __attribute__((ext_vector_type(4)))  __bf16 v4bf;
typedef __attribute__((ext_vector_type(8)))  float  v8f;

#define D_DIM   1024
#define DR      256
#define NSTAGE  4
#define GSZ     8
#define NGRP    128
#define PWIN    40      // window width (one half of y)
#define KPAD    96      // 80 -> padded to 3 * 32
#define TILE_M  16
#define WAVES   8
#define NTHREADS (WAVES * 32)

// LDS layout (bytes)
#define OFF_XS  0                               // [16][1024] bf16 = 32768
#define OFF_Z0  (OFF_XS + TILE_M * D_DIM * 2)   // [16][1024] bf16 = 32768
#define OFF_Z1  (OFF_Z0 + TILE_M * D_DIM * 2)   // [16][1024] bf16 = 32768
#define OFF_WI  (OFF_Z1 + TILE_M * D_DIM * 2)   // [256][96]  bf16 = 49152
#define OFF_WO  (OFF_WI + DR * KPAD * 2)        // [16][256]  bf16 = 8192
#define OFF_BI  (OFF_WO + 16 * DR * 2)          // [256]      f32  = 1024
#define OFF_HS  (OFF_BI + DR * 4)               // 8 x [16][256] bf16 = 65536
#define SMEM_TOTAL (OFF_HS + WAVES * TILE_M * DR * 2)  // = 222208 bytes

union Frag16 { v16bf v; v8bf h[2]; };

// Branch-free tanh: sign(x) * (1 - e^(-2|x|)) / (1 + e^(-2|x|)).
// ~6 VALU + 2 TRANS (v_exp_f32, v_rcp_f32), no EXEC manipulation; TRANS co-executes
// with XDL WMMA per the CDNA5 scheduling model.
__device__ __forceinline__ float fast_tanh(float x) {
  float ax = __builtin_fabsf(x);
  float e  = __builtin_amdgcn_exp2f(ax * -2.8853900817779268f); // -2*log2(e)
  float r  = (1.0f - e) * __builtin_amdgcn_rcpf(1.0f + e);
  return __builtin_copysignf(r, x);
}

// One 8-element K-run of a GEMM1 A-fragment: logical y-column c0 (multiple of
// 8, < 80) maps to the x-window (c0 < 40) or z-window (c0 >= 40); runs never
// cross the mod-1024 wrap, so each is a single contiguous ds_load_b128 half.
__device__ __forceinline__ v8bf load_y_run(const __bf16* xs, const __bf16* zs,
                                           int row, int base, int c0) {
  const __bf16* src = (c0 < PWIN)
      ? (xs + row * D_DIM + ((base + c0) & (D_DIM - 1)))
      : (zs + row * D_DIM + ((base + c0 - PWIN) & (D_DIM - 1)));
  return *reinterpret_cast<const v8bf*>(src);
}

__global__ void local_deep_rfm_kernel(const float* __restrict__ x,
                                      const float* __restrict__ Wi,
                                      const float* __restrict__ bi,
                                      const float* __restrict__ Wo,
                                      float* __restrict__ out,
                                      int nrows) {
  extern __shared__ char smem[];
  __bf16* xs   = reinterpret_cast<__bf16*>(smem + OFF_XS);   // [16][1024]
  __bf16* z0   = reinterpret_cast<__bf16*>(smem + OFF_Z0);   // [16][1024]
  __bf16* z1   = reinterpret_cast<__bf16*>(smem + OFF_Z1);   // [16][1024]
  __bf16* wi_s = reinterpret_cast<__bf16*>(smem + OFF_WI);   // [256][96]
  __bf16* wo_s = reinterpret_cast<__bf16*>(smem + OFF_WO);   // [16][256]
  float*  bi_s = reinterpret_cast<float*>(smem + OFF_BI);    // [256]

  const int tid   = threadIdx.x;
  const int w     = tid >> 5;          // wave id 0..7
  const int lane  = tid & 31;
  const int lrow  = lane & 15;         // A-row / B-col within 16-tile
  const int khalf = (lane >> 4) & 1;
  const int row0  = blockIdx.x * TILE_M;

  __bf16* hsw = reinterpret_cast<__bf16*>(smem + OFF_HS) + w * (TILE_M * DR); // [16][256]

  // ---- stage-in x tile as bf16; z0 := x (vectorized: float4 -> 4 x bf16) --
  for (int t = tid; t < TILE_M * (D_DIM / 4); t += NTHREADS) {
    int r  = t / (D_DIM / 4);
    int c4 = t % (D_DIM / 4);
    int row = row0 + r;
    float4 v = make_float4(0.f, 0.f, 0.f, 0.f);
    if (row < nrows)
      v = reinterpret_cast<const float4*>(x)[(size_t)row * (D_DIM / 4) + c4];
    v4bf bv = {(__bf16)v.x, (__bf16)v.y, (__bf16)v.z, (__bf16)v.w};
    *reinterpret_cast<v4bf*>(xs + r * D_DIM + c4 * 4) = bv;
    *reinterpret_cast<v4bf*>(z0 + r * D_DIM + c4 * 4) = bv;
  }

#pragma unroll 1
  for (int stage = 0; stage < NSTAGE; ++stage) {
    const __bf16* zprev = (stage & 1) ? z1 : z0;
    __bf16*       znext = (stage & 1) ? z0 : z1;

    __syncthreads();  // prev-stage z writes complete; weight LDS reusable

    // ---- stage weights into LDS (bf16, K padded to 96) --------------------
    for (int t = tid; t < DR * (2 * PWIN) / 4; t += NTHREADS) {
      int r = t / (2 * PWIN / 4), p4 = t % (2 * PWIN / 4);
      float4 v = reinterpret_cast<const float4*>(
          Wi + (size_t)stage * DR * 2 * PWIN)[r * (2 * PWIN / 4) + p4];
      v4bf bv = {(__bf16)v.x, (__bf16)v.y, (__bf16)v.z, (__bf16)v.w};
      *reinterpret_cast<v4bf*>(wi_s + r * KPAD + p4 * 4) = bv;
    }
    for (int t = tid; t < DR * 4; t += NTHREADS) {     // pad cols 80..95
      int r = t / 4, p = 2 * PWIN + (t % 4) * 4;
      v4bf zv = {(__bf16)0.f, (__bf16)0.f, (__bf16)0.f, (__bf16)0.f};
      *reinterpret_cast<v4bf*>(wi_s + r * KPAD + p) = zv;
    }
    for (int t = tid; t < GSZ * DR / 4; t += NTHREADS) {
      float4 v = reinterpret_cast<const float4*>(
          Wo + (size_t)stage * GSZ * DR)[t];
      v4bf bv = {(__bf16)v.x, (__bf16)v.y, (__bf16)v.z, (__bf16)v.w};
      v4bf zv = {(__bf16)0.f, (__bf16)0.f, (__bf16)0.f, (__bf16)0.f};
      *reinterpret_cast<v4bf*>(wo_s + t * 4) = bv;            // cols 0..7
      *reinterpret_cast<v4bf*>(wo_s + GSZ * DR + t * 4) = zv; // cols 8..15 pad
    }
    for (int t = tid; t < DR; t += NTHREADS) bi_s[t] = bi[stage * DR + t];

    // prefetch next stage's Wi into GL2 while we compute this stage
    if (stage + 1 < NSTAGE) {
      const float* nw = Wi + (size_t)(stage + 1) * DR * 2 * PWIN;
      for (int t = tid * 16; t < DR * 2 * PWIN; t += NTHREADS * 16)
        __builtin_prefetch(nw + t, 0, 1);
    }
    __syncthreads();

    // ---- each wave processes 16 groups ------------------------------------
#pragma unroll 1
    for (int gi = 0; gi < NGRP / WAVES; ++gi) {
      const int g    = w * (NGRP / WAVES) + gi;
      const int base = g * GSZ - 2 * GSZ + D_DIM;   // window start (mod D)
      const int cA   = 8 * khalf;

      // A fragments (y = [x_win(40) | z_win(40) | 0-pad(16)]) straight from LDS
      Frag16 fa0, fa1, fa2;
      fa0.h[0] = load_y_run(xs, zprev, lrow, base, cA + 0);
      fa0.h[1] = load_y_run(xs, zprev, lrow, base, cA + 16);
      fa1.h[0] = load_y_run(xs, zprev, lrow, base, cA + 32);
      fa1.h[1] = load_y_run(xs, zprev, lrow, base, cA + 48);
      fa2.h[0] = load_y_run(xs, zprev, lrow, base, cA + 64);
      fa2.h[1] = (v8bf){(__bf16)0.f, (__bf16)0.f, (__bf16)0.f, (__bf16)0.f,
                        (__bf16)0.f, (__bf16)0.f, (__bf16)0.f, (__bf16)0.f};

      // GEMM1: h[16][256] = y[16][96] x Wi^T[96][256]; bias + tanh -> LDS
#pragma unroll 1
      for (int rt = 0; rt < DR / 16; ++rt) {
        const __bf16* wrow = wi_s + (rt * 16 + lrow) * KPAD + cA;
        Frag16 b0, b1, b2;
        b0.h[0] = *reinterpret_cast<const v8bf*>(wrow + 0);
        b0.h[1] = *reinterpret_cast<const v8bf*>(wrow + 16);
        b1.h[0] = *reinterpret_cast<const v8bf*>(wrow + 32);
        b1.h[1] = *reinterpret_cast<const v8bf*>(wrow + 48);
        b2.h[0] = *reinterpret_cast<const v8bf*>(wrow + 64);
        b2.h[1] = *reinterpret_cast<const v8bf*>(wrow + 80);

        v8f acc = {0.f, 0.f, 0.f, 0.f, 0.f, 0.f, 0.f, 0.f};
        acc = __builtin_amdgcn_wmma_f32_16x16x32_bf16(
            false, fa0.v, false, b0.v, (short)0, acc, false, false);
        acc = __builtin_amdgcn_wmma_f32_16x16x32_bf16(
            false, fa1.v, false, b1.v, (short)0, acc, false, false);
        acc = __builtin_amdgcn_wmma_f32_16x16x32_bf16(
            false, fa2.v, false, b2.v, (short)0, acc, false, false);

        const int ncol = rt * 16 + lrow;
        const float bv = bi_s[ncol];
#pragma unroll
        for (int j = 0; j < 8; ++j) {
          int m = j + 8 * khalf;                    // C-layout row
          hsw[m * DR + ncol] = (__bf16)fast_tanh(acc[j] + bv);
        }
      }
      asm volatile("s_wait_dscnt 0" ::: "memory");

      // GEMM2: z_g[16][8] = h[16][256] x Wo^T[256][8(pad16)]
      v8f acc2 = {0.f, 0.f, 0.f, 0.f, 0.f, 0.f, 0.f, 0.f};
#pragma unroll
      for (int kt = 0; kt < DR / 32; ++kt) {
        const __bf16* ha = hsw + lrow * DR + kt * 32 + cA;
        const __bf16* wb = wo_s + lrow * DR + kt * 32 + cA;
        Frag16 a, b;
        a.h[0] = *reinterpret_cast<const v8bf*>(ha);
        a.h[1] = *reinterpret_cast<const v8bf*>(ha + 16);
        b.h[0] = *reinterpret_cast<const v8bf*>(wb);
        b.h[1] = *reinterpret_cast<const v8bf*>(wb + 16);
        acc2 = __builtin_amdgcn_wmma_f32_16x16x32_bf16(
            false, a.v, false, b.v, (short)0, acc2, false, false);
      }
      if (lrow < GSZ) {
        const int zcol = g * GSZ + lrow;
#pragma unroll
        for (int j = 0; j < 8; ++j) {
          int m = j + 8 * khalf;
          znext[m * D_DIM + zcol] = (__bf16)acc2[j];   // next-stage feedback
          if (stage == NSTAGE - 1) {
            int row = row0 + m;
            if (row < nrows) out[(size_t)row * D_DIM + zcol] = acc2[j];
          }
        }
      }
    }
  }
}

extern "C" void kernel_launch(void* const* d_in, const int* in_sizes, int n_in,
                              void* d_out, int out_size, void* d_ws, size_t ws_size,
                              hipStream_t stream) {
  const float* x  = (const float*)d_in[0];   // [N, 1024]
  const float* Wi = (const float*)d_in[1];   // [4, 256, 80]
  const float* bi = (const float*)d_in[2];   // [4, 256]
  const float* Wo = (const float*)d_in[3];   // [4, 8, 256]
  float* out = (float*)d_out;                // [N, 1024]

  const int nrows = in_sizes[0] / D_DIM;
  const int grid  = (nrows + TILE_M - 1) / TILE_M;

  local_deep_rfm_kernel<<<grid, NTHREADS, SMEM_TOTAL, stream>>>(
      x, Wi, bi, Wo, out, nrows);
}
